// MultiHeadAttention_85401129713922
// MI455X (gfx1250) — matrix operand order repaired and checked by reading, not compiled, not run on hardware
//
#include <hip/hip_runtime.h>
#include <hip/hip_bf16.h>

typedef __attribute__((ext_vector_type(16))) _Float16 v16h;
typedef __attribute__((ext_vector_type(8)))  _Float16 v8h;
typedef __attribute__((ext_vector_type(8)))  float    v8f;

#define BATCH 2
#define SEQ   2048
#define EMB   512
#define HEADS 8
#define HD    64
#define QKV3  1536
#define MTOT  (BATCH*SEQ)      // 4096
#define HALFWIN 128
#define KMAX  320              // max banded keys per 16-query tile (padded)

// ---------------- WMMA helpers (layouts per CDNA5 ISA 7.12.2) ----------------

__device__ __forceinline__ v8f wmma16(v16h a, v16h b, v8f c) {
  return __builtin_amdgcn_wmma_f32_16x16x32_f16(
      false, a, false, b, (short)0, c, false, false);
}

// A operand: 16x32 f16, source row-major [M x K], row stride lda (halfs).
// lane lo = M row; elements 0..7 = K k0+8*hi .. +7 ; elements 8..15 = +16 halfs.
__device__ __forceinline__ v16h load_frag_a(const _Float16* base, int lda,
                                            int m, int k0, int lane) {
  const int hi = lane >> 4, lo = lane & 15;
  const _Float16* p = base + (size_t)(m + lo) * lda + k0 + hi * 8;
  v16h a;
  ((v8h*)&a)[0] = *(const v8h*)(p);
  ((v8h*)&a)[1] = *(const v8h*)(p + 16);
  return a;
}

// B operand: 32x16 f16, source is "N-major" [N x K], row stride ldb (halfs):
// lane lo = N col; elements 0..7 = K k0+8*hi .. +7 ; elements 8..15 = +16 halfs.
__device__ __forceinline__ v16h load_frag_b(const _Float16* base, int ldb,
                                            int n, int k0, int lane) {
  const int hi = lane >> 4, lo = lane & 15;
  const _Float16* p = base + (size_t)(n + lo) * ldb + k0 + hi * 8;
  v16h b;
  ((v8h*)&b)[0] = *(const v8h*)(p);
  ((v8h*)&b)[1] = *(const v8h*)(p + 16);
  return b;
}

// Async DMA: stage `halfs8` 16-byte chunks from global `src` into shared `dst`,
// cooperatively across the block, using GLOBAL_LOAD_ASYNC_TO_LDS_B128
// (ASYNCcnt-tracked, no VGPR round trip). Caller must __syncthreads() after.
__device__ __forceinline__ void async_stage_a(const _Float16* __restrict__ src,
                                              _Float16* dst_lds, int nchunks,
                                              int tid, int nthreads) {
  for (int c = tid; c < nchunks; c += nthreads) {
    unsigned lds_addr =
        (unsigned)(unsigned long long)(dst_lds + (size_t)c * 8);
    unsigned long long gaddr =
        (unsigned long long)(src + (size_t)c * 8);
    asm volatile("global_load_async_to_lds_b128 %0, %1, off"
                 :: "v"(lds_addr), "v"(gaddr) : "memory");
  }
  asm volatile("s_wait_asynccnt 0x0" ::: "memory");
}

// ---------------- prep kernels ----------------

__global__ __launch_bounds__(256) void cvt_f32_f16(const float* __restrict__ in,
                                                   _Float16* __restrict__ out, int n) {
  for (int i = blockIdx.x * blockDim.x + threadIdx.x; i < n;
       i += gridDim.x * blockDim.x)
    out[i] = (_Float16)in[i];
}

// in: [K x N] f32 row-major -> out: [N x K] f16 row-major (N-major)
__global__ __launch_bounds__(256) void transpose_cvt(const float* __restrict__ in,
                                                     _Float16* __restrict__ out,
                                                     int K, int N) {
  int total = K * N;
  for (int i = blockIdx.x * blockDim.x + threadIdx.x; i < total;
       i += gridDim.x * blockDim.x) {
    int n = i / K, k = i - n * K;
    out[i] = (_Float16)in[(size_t)k * N + n];
  }
}

// ---------------- QKV GEMM: [4096x512] @ [512x1536] + bias, scatter to Q/K/Vt
// A panel (64 x 512 = 64KB) staged to LDS via async DMA, shared by all 8 waves.
// Each wave computes 32(M) x 64(N): 8 wmma per k-step off 2 LDS + 4 global frags.

__global__ __launch_bounds__(256) void qkv_gemm(const _Float16* __restrict__ xh,
                                                const _Float16* __restrict__ wT,
                                                const float* __restrict__ bias,
                                                _Float16* __restrict__ Qh,
                                                _Float16* __restrict__ Kh,
                                                _Float16* __restrict__ Vt) {
  __shared__ __align__(16) _Float16 As[64 * EMB];   // 64 KB

  const int lane = threadIdx.x & 31;
  const int wave = threadIdx.x >> 5;
  const int rowM = blockIdx.y * 64;                 // block's A rows
  const int tileMl = (wave & 1) * 32;               // LDS-local M
  const int tileN  = blockIdx.x * 256 + (wave >> 1) * 64;

  async_stage_a(xh + (size_t)rowM * EMB, As, 64 * EMB / 8,
                threadIdx.x, 256);
  __syncthreads();

  v8f acc[2][4];
#pragma unroll
  for (int mi = 0; mi < 2; ++mi)
#pragma unroll
    for (int ni = 0; ni < 4; ++ni) acc[mi][ni] = (v8f){};

  for (int k0 = 0; k0 < EMB; k0 += 32) {
    v16h a0 = load_frag_a(As, EMB, tileMl,      k0, lane);
    v16h a1 = load_frag_a(As, EMB, tileMl + 16, k0, lane);
#pragma unroll
    for (int ni = 0; ni < 4; ++ni) {
      v16h b = load_frag_b(wT, EMB, tileN + ni * 16, k0, lane);
      acc[0][ni] = wmma16(a0, b, acc[0][ni]);
      acc[1][ni] = wmma16(a1, b, acc[1][ni]);
    }
  }

  const int lo = lane & 15, hi = lane >> 4;
#pragma unroll
  for (int ni = 0; ni < 4; ++ni) {
    const int n = tileN + ni * 16 + lo;
    const float bn = bias[n];
    const int h = n / 192, cc = n - h * 192;
#pragma unroll
    for (int mi = 0; mi < 2; ++mi) {
#pragma unroll
      for (int r = 0; r < 8; ++r) {
        const int m = rowM + tileMl + mi * 16 + r + 8 * hi;
        const int b = m >> 11, s = m & (SEQ - 1);
        const _Float16 hv = (_Float16)(acc[mi][ni][r] + bn);
        if (cc < 64)
          Qh[(((size_t)(b * HEADS + h) * SEQ) + s) * HD + cc] = hv;
        else if (cc < 128)
          Kh[(((size_t)(b * HEADS + h) * SEQ) + s) * HD + (cc - 64)] = hv;
        else
          Vt[(((size_t)(b * HEADS + h) * HD) + (cc - 128)) * SEQ + s] = hv;
      }
    }
  }
}

// ---------------- banded attention: one wave per (b, h, 16-query tile) ----------------

__global__ __launch_bounds__(32) void attn_band(const _Float16* __restrict__ Qh,
                                                const _Float16* __restrict__ Kh,
                                                const _Float16* __restrict__ Vt,
                                                const int* __restrict__ pmask,
                                                _Float16* __restrict__ valsH) {
  __shared__ __align__(16) float    sc[16 * KMAX];
  __shared__ __align__(16) _Float16 at[16 * KMAX];
  __shared__ float rinv[16];
  __shared__ int   pmq[16];

  const int lane = threadIdx.x;
  const int lo = lane & 15, hi = lane >> 4;
  const int idx = blockIdx.x;
  const int mt = idx & (SEQ / 16 - 1);
  const int h  = (idx >> 7) & (HEADS - 1);
  const int b  = idx >> 10;
  const int m0 = mt * 16;

  const _Float16* Qb = Qh + (size_t)(b * HEADS + h) * SEQ * HD;
  const _Float16* Kb = Kh + (size_t)(b * HEADS + h) * SEQ * HD;
  const _Float16* Vb = Vt + (size_t)(b * HEADS + h) * HD * SEQ;

  int klow = m0 - HALFWIN; if (klow < 0) klow = 0;
  const int kbase = klow & ~31;
  int kend = m0 + 16 + HALFWIN; if (kend > SEQ) kend = SEQ;
  const int nkt   = (kend - kbase + 15) >> 4;   // 16-key tiles
  const int nkt32 = (nkt + 1) >> 1;             // 32-key chunks for PV

  if (lane < 16) pmq[lo] = pmask[b * SEQ + m0 + lo];
  __syncthreads();

  // Q fragments (16 rows x 64 dims)
  const v16h q0 = load_frag_a(Qb, HD, m0, 0, lane);
  const v16h q1 = load_frag_a(Qb, HD, m0, 32, lane);

  // ---- scores: Q K^T, masked to -inf, *1/sqrt(hd) ----
  for (int t = 0; t < nkt; ++t) {
    const int kb = kbase + t * 16;
    v8f d = {};
    d = wmma16(q0, load_frag_b(Kb, HD, kb, 0, lane), d);
    d = wmma16(q1, load_frag_b(Kb, HD, kb, 32, lane), d);
    const int key = kb + lo;
    const bool pk = pmask[b * SEQ + key] != 0;
#pragma unroll
    for (int r = 0; r < 8; ++r) {
      const int m = r + 8 * hi;
      int diff = (m0 + m) - key; if (diff < 0) diff = -diff;
      const bool valid = (diff <= HALFWIN) && pk && (pmq[m] != 0);
      sc[m * KMAX + (kb - kbase) + lo] =
          valid ? d[r] * 0.125f : -__builtin_inff();
    }
  }
  __syncthreads();

  // ---- row softmax (unnormalized exp; 1/sum folded into PV epilogue) ----
  const int nk = nkt * 16;
  if (lane < 16) {
    const int m = lane;
    float mx = -__builtin_inff();
    for (int k = 0; k < nk; ++k) {
      float v = sc[m * KMAX + k];
      mx = v > mx ? v : mx;
    }
    if (mx == -__builtin_inff()) {
      for (int k = 0; k < nk; ++k) at[m * KMAX + k] = (_Float16)0.f;
      rinv[m] = 0.f;
    } else {
      float sum = 0.f;
      for (int k = 0; k < nk; ++k) {
        float e = __expf(sc[m * KMAX + k] - mx);
        sum += e;
        at[m * KMAX + k] = (_Float16)e;
      }
      rinv[m] = sum > 0.f ? 1.f / sum : 0.f;
    }
    for (int k = nk; k < nkt32 * 32; ++k) at[m * KMAX + k] = (_Float16)0.f;
  }
  __syncthreads();

  // ---- PV: attn [16 x nk] @ V [nk x 64] ----
  v8f o0 = {}, o1 = {}, o2 = {}, o3 = {};
  for (int t = 0; t < nkt32; ++t) {
    const int kk = t * 32;
    const int kb = kbase + kk;
    const v16h a = load_frag_a(at, KMAX, 0, kk, lane);
    o0 = wmma16(a, load_frag_b(Vb, SEQ, 0,  kb, lane), o0);
    o1 = wmma16(a, load_frag_b(Vb, SEQ, 16, kb, lane), o1);
    o2 = wmma16(a, load_frag_b(Vb, SEQ, 32, kb, lane), o2);
    o3 = wmma16(a, load_frag_b(Vb, SEQ, 48, kb, lane), o3);
  }

  const v8f* frags[4] = { &o0, &o1, &o2, &o3 };
#pragma unroll
  for (int f = 0; f < 4; ++f) {
    const int d0 = f * 16;
#pragma unroll
    for (int r = 0; r < 8; ++r) {
      const int m = r + 8 * hi;
      const float val = (*frags[f])[r] * rinv[m];
      valsH[((size_t)(b * SEQ + m0 + m)) * EMB + h * HD + d0 + lo] =
          (_Float16)val;
    }
  }
}

// ---------------- output GEMM: [4096x512] @ [512x512] + bo -> fp32
// Same async-LDS A staging + 32x64-per-wave register blocking.

__global__ __launch_bounds__(256) void out_gemm(const _Float16* __restrict__ vh,
                                                const _Float16* __restrict__ wT,
                                                const float* __restrict__ bias,
                                                float* __restrict__ out) {
  __shared__ __align__(16) _Float16 As[64 * EMB];   // 64 KB

  const int lane = threadIdx.x & 31;
  const int wave = threadIdx.x >> 5;
  const int rowM = blockIdx.y * 64;
  const int tileMl = (wave & 1) * 32;
  const int tileN  = blockIdx.x * 256 + (wave >> 1) * 64;

  async_stage_a(vh + (size_t)rowM * EMB, As, 64 * EMB / 8,
                threadIdx.x, 256);
  __syncthreads();

  v8f acc[2][4];
#pragma unroll
  for (int mi = 0; mi < 2; ++mi)
#pragma unroll
    for (int ni = 0; ni < 4; ++ni) acc[mi][ni] = (v8f){};

  for (int k0 = 0; k0 < EMB; k0 += 32) {
    v16h a0 = load_frag_a(As, EMB, tileMl,      k0, lane);
    v16h a1 = load_frag_a(As, EMB, tileMl + 16, k0, lane);
#pragma unroll
    for (int ni = 0; ni < 4; ++ni) {
      v16h b = load_frag_b(wT, EMB, tileN + ni * 16, k0, lane);
      acc[0][ni] = wmma16(a0, b, acc[0][ni]);
      acc[1][ni] = wmma16(a1, b, acc[1][ni]);
    }
  }

  const int lo = lane & 15, hi = lane >> 4;
#pragma unroll
  for (int ni = 0; ni < 4; ++ni) {
    const int n = tileN + ni * 16 + lo;
    const float bn = bias[n];
#pragma unroll
    for (int mi = 0; mi < 2; ++mi) {
#pragma unroll
      for (int r = 0; r < 8; ++r) {
        const int m = rowM + tileMl + mi * 16 + r + 8 * hi;
        out[(size_t)m * EMB + n] = acc[mi][ni][r] + bn;
      }
    }
  }
}

// ---------------- host launch ----------------

extern "C" void kernel_launch(void* const* d_in, const int* in_sizes, int n_in,
                              void* d_out, int out_size, void* d_ws, size_t ws_size,
                              hipStream_t stream) {
  const float* x     = (const float*)d_in[0];
  const int*   pmask = (const int*)  d_in[1];
  const float* Wqkv  = (const float*)d_in[2];   // [512 x 1536]
  const float* bqkv  = (const float*)d_in[3];
  const float* Wo    = (const float*)d_in[4];   // [512 x 512]
  const float* bo    = (const float*)d_in[5];
  float* out = (float*)d_out;

  char* ws = (char*)d_ws;
  size_t off = 0;
  auto alloc = [&](size_t bytes) {
    char* p = ws + off;
    off += (bytes + 255) & ~(size_t)255;
    return p;
  };
  _Float16* xh    = (_Float16*)alloc((size_t)MTOT * EMB * 2);
  _Float16* wqkvT = (_Float16*)alloc((size_t)QKV3 * EMB * 2);
  _Float16* woT   = (_Float16*)alloc((size_t)EMB * EMB * 2);
  _Float16* Qh    = (_Float16*)alloc((size_t)BATCH * HEADS * SEQ * HD * 2);
  _Float16* Kh    = (_Float16*)alloc((size_t)BATCH * HEADS * SEQ * HD * 2);
  _Float16* Vt    = (_Float16*)alloc((size_t)BATCH * HEADS * HD * SEQ * 2);
  _Float16* valsH = (_Float16*)alloc((size_t)MTOT * EMB * 2);
  (void)ws_size; (void)n_in; (void)in_sizes; (void)out_size;

  cvt_f32_f16<<<1024, 256, 0, stream>>>(x, xh, MTOT * EMB);
  transpose_cvt<<<1024, 256, 0, stream>>>(Wqkv, wqkvT, EMB, QKV3);
  transpose_cvt<<<512, 256, 0, stream>>>(Wo, woT, EMB, EMB);

  qkv_gemm<<<dim3(QKV3 / 256, MTOT / 64), 256, 0, stream>>>(
      xh, wqkvT, bqkv, Qh, Kh, Vt);

  attn_band<<<BATCH * HEADS * (SEQ / 16), 32, 0, stream>>>(
      Qh, Kh, Vt, pmask, valsH);

  out_gemm<<<dim3(EMB / 256, MTOT / 64), 256, 0, stream>>>(
      valsH, woT, bo, out);
}
